// VectorQuantizer_normal_17841294148022
// MI455X (gfx1250) — matrix operand order repaired and checked by reading, not compiled, not run on hardware
//
#include <hip/hip_runtime.h>
#include <hip/hip_bf16.h>

// Vector-Quantizer forward for MI455X (gfx1250, wave32, bf16 WMMA 16x16x32).
//
// Pipeline (all on `stream`):
//   1. cvt_inputs : f32 [N,1000] -> bf16 [N,1024] (zero-padded)
//   2. cvt_emb    : f32 [K,1000] -> bf16 [K,1024], e_norm[k] = ||e_k||^2, hist[k]=0
//   3. vq_argmin  : fused bf16-WMMA GEMM (x . E^T) + running argmin over K.
//                   A (16 rows/wave) is loaded ONCE into an explicit register
//                   array (32 fragments = 256 VGPRs, high-VGPR space); B (2MB
//                   bf16 codebook) streams from L2 with the next tile prefetched
//                   at WGP scope into WGP$ while the current tile's WMMAs run.
//   4. vq_epilogue: per row - one-hot encodings, quantized_st = x + (e[idx]-x),
//                   row-wise sum of (e[idx]-x)^2, histogram count.
//                   Output stores are non-temporal so the 132MB of streaming
//                   writes don't evict the L2-resident inputs/codebook.
//   5. vq_final   : deterministic tree-reductions -> loss, perplexity

#define N_  16384
#define D_  1000
#define DP  1024
#define K_  1024

typedef __attribute__((ext_vector_type(16))) __bf16 v16bf;
typedef __attribute__((ext_vector_type(8)))  float  v8f;
typedef __attribute__((ext_vector_type(4)))  int    v4i;

// ---------------- 1. convert inputs to padded bf16 ----------------
__global__ __launch_bounds__(256) void cvt_inputs(const float* __restrict__ x,
                                                  __bf16* __restrict__ xb) {
    size_t i = (size_t)blockIdx.x * 256 + threadIdx.x;   // over N*DP
    int d = (int)(i & (DP - 1));
    size_t n = i >> 10;
    float v = (d < D_) ? x[n * D_ + d] : 0.0f;
    xb[i] = (__bf16)v;
}

// ---------------- 2. convert emb, compute ||e_k||^2, zero hist ----------------
__global__ __launch_bounds__(256) void cvt_emb(const float* __restrict__ e,
                                               __bf16* __restrict__ eb,
                                               float* __restrict__ enorm,
                                               int* __restrict__ hist) {
    int k = blockIdx.x;
    int tid = threadIdx.x;
    float s = 0.0f;
    for (int d = tid; d < DP; d += 256) {
        float v = (d < D_) ? e[(size_t)k * D_ + d] : 0.0f;
        eb[(size_t)k * DP + d] = (__bf16)v;
        s += v * v;
    }
    __shared__ float red[256];
    red[tid] = s;
    __syncthreads();
    for (int o = 128; o > 0; o >>= 1) {
        if (tid < o) red[tid] += red[tid + o];
        __syncthreads();
    }
    if (tid == 0) {
        enorm[k] = red[0];
        hist[k]  = 0;
    }
}

// ---------------- 3. fused bf16 WMMA GEMM + argmin ----------------
// blockDim = 128 (4 waves). Each wave owns 16 rows, sweeps all 64 column tiles.
__global__ __launch_bounds__(128) void vq_argmin(const __bf16* __restrict__ xb,
                                                 const __bf16* __restrict__ eb,
                                                 const float* __restrict__ enorm,
                                                 int* __restrict__ outIdx) {
    const int lane = threadIdx.x & 31;
    const int wave = threadIdx.x >> 5;
    const int l    = lane & 15;      // row/col within tile
    const int hi   = lane >> 4;      // half-wave select
    const int rowBase = (blockIdx.x * 4 + wave) * 16;

    const __bf16* arow = xb + (size_t)(rowBase + l) * DP;

    // Load ALL A fragments up front: loop-invariant across the 64 column
    // tiles, so they live in registers for the whole kernel (32 x 8 VGPRs,
    // constant indices after full unroll -> no scratch spill).
    // A (16x32 bf16): lane holds K = {hi*8 .. +7} and {16+hi*8 .. +7}.
    union Frag { v4i q[2]; v16bf v; };
    Frag A[32];
#pragma unroll
    for (int dk = 0; dk < 32; ++dk) {
        A[dk].q[0] = *(const v4i*)(arow + dk * 32 + hi * 8);
        A[dk].q[1] = *(const v4i*)(arow + dk * 32 + 16 + hi * 8);
    }

    float bestV[8];
    int   bestI[8];
#pragma unroll
    for (int r = 0; r < 8; ++r) { bestV[r] = 3.4e38f; bestI[r] = 0; }

    for (int ct = 0; ct < K_ / 16; ++ct) {
        const int colBase = ct * 16;
        const __bf16* brow = eb + (size_t)(colBase + l) * DP;

        // Prefetch the NEXT column tile's B block (contiguous 32KB) into the
        // WGP$ (SCOPE=0: "pull into all cache levels") while this tile's 32
        // WMMAs execute. 32 lanes x 128B stride x 8 offsets = 32KB. Addresses
        // are always valid inside eb, so non-speculative WGP-scope is safe.
        {
            const int nextBase = ((ct + 1) & 63) * 16;
            const char* nb = (const char*)(eb + (size_t)nextBase * DP) + lane * 128;
            asm volatile(
                "global_prefetch_b8 %0, off\n\t"
                "global_prefetch_b8 %0, off offset:4096\n\t"
                "global_prefetch_b8 %0, off offset:8192\n\t"
                "global_prefetch_b8 %0, off offset:12288\n\t"
                "global_prefetch_b8 %0, off offset:16384\n\t"
                "global_prefetch_b8 %0, off offset:20480\n\t"
                "global_prefetch_b8 %0, off offset:24576\n\t"
                "global_prefetch_b8 %0, off offset:28672"
                :: "v"(nb));
        }

        v8f acc = {};
        // B (32x16 bf16): lane holds K = {hi*8 .. +7} and {16+hi*8 .. +7} of column l (= emb row)
#pragma unroll
        for (int dk = 0; dk < 32; ++dk) {
            Frag B;
            B.q[0] = *(const v4i*)(brow + dk * 32 + hi * 8);
            B.q[1] = *(const v4i*)(brow + dk * 32 + 16 + hi * 8);
            acc = __builtin_amdgcn_wmma_f32_16x16x32_bf16(
                false, A[dk].v, false, B.v, (short)0, acc, false, false);
        }

        // score = ||e||^2 - 2 * dot ; column for this lane = colBase + l
        const int col = colBase + l;
        const float en = enorm[col];
#pragma unroll
        for (int r = 0; r < 8; ++r) {
            float s = en - 2.0f * acc[r];
            if (s < bestV[r]) { bestV[r] = s; bestI[r] = col; }
        }
    }

    // C/D layout: lanes 0-15 hold rows 0..7 (VGPR r -> M=r), lanes 16-31 rows 8..15.
    // Reduce argmin across the 16 lanes of each half (xor masks stay within half).
#pragma unroll
    for (int r = 0; r < 8; ++r) {
        float v = bestV[r];
        int   i = bestI[r];
        for (int off = 1; off < 16; off <<= 1) {
            float ov = __shfl_xor(v, off, 32);
            int   oi = __shfl_xor(i, off, 32);
            if (ov < v || (ov == v && oi < i)) { v = ov; i = oi; }
        }
        bestV[r] = v;
        bestI[r] = i;
    }
    if (l == 0) {
#pragma unroll
        for (int r = 0; r < 8; ++r)
            outIdx[rowBase + hi * 8 + r] = bestI[r];
    }
}

// ---------------- 4. epilogue: one block per row ----------------
__global__ __launch_bounds__(256) void vq_epilogue(const float* __restrict__ x,
                                                   const float* __restrict__ e,
                                                   const int* __restrict__ idxArr,
                                                   float* __restrict__ outQ,
                                                   float* __restrict__ outEnc,
                                                   float* __restrict__ partial,
                                                   int* __restrict__ hist) {
    const int n   = blockIdx.x;
    const int tid = threadIdx.x;
    const int idx = idxArr[n];

    // one-hot encodings row: pure streaming write -> non-temporal
    float* enc = outEnc + (size_t)n * K_;
    for (int j = tid; j < K_; j += 256)
        __builtin_nontemporal_store((j == idx) ? 1.0f : 0.0f, enc + j);

    const float* xr = x + (size_t)n * D_;
    const float* er = e + (size_t)idx * D_;
    float* qr = outQ + (size_t)n * D_;

    float s = 0.0f;
    for (int d = tid; d < D_; d += 256) {
        float xv = xr[d];
        float qv = er[d];
        __builtin_nontemporal_store(xv + (qv - xv), qr + d);  // straight-through
        float df = qv - xv;
        s += df * df;
    }
    __shared__ float red[256];
    red[tid] = s;
    __syncthreads();
    for (int o = 128; o > 0; o >>= 1) {
        if (tid < o) red[tid] += red[tid + o];
        __syncthreads();
    }
    if (tid == 0) {
        partial[n] = red[0];
        atomicAdd(&hist[idx], 1);      // integer -> deterministic
    }
}

// ---------------- 5. final: loss + perplexity ----------------
__global__ __launch_bounds__(1024) void vq_final(const float* __restrict__ partial,
                                                 const int* __restrict__ hist,
                                                 float* __restrict__ out) {
    const int tid = threadIdx.x;   // 1024
    __shared__ float red[1024];

    // deterministic fixed-order sum of 16384 row partials
    float s = 0.0f;
    for (int i = tid; i < N_; i += 1024) s += partial[i];
    red[tid] = s;
    __syncthreads();
    for (int o = 512; o > 0; o >>= 1) {
        if (tid < o) red[tid] += red[tid + o];
        __syncthreads();
    }
    float loss = 1.25f * red[0] / (float)((size_t)N_ * D_);
    __syncthreads();

    // entropy over K=1024 bins (one bin per thread)
    float p = (float)hist[tid] / (float)N_;
    red[tid] = p * logf(p + 1e-10f);
    __syncthreads();
    for (int o = 512; o > 0; o >>= 1) {
        if (tid < o) red[tid] += red[tid + o];
        __syncthreads();
    }
    if (tid == 0) {
        out[0] = loss;                                  // loss
        out[1 + (size_t)N_ * D_] = expf(-red[0]);       // perplexity
    }
}

// ---------------- launch ----------------
extern "C" void kernel_launch(void* const* d_in, const int* in_sizes, int n_in,
                              void* d_out, int out_size, void* d_ws, size_t ws_size,
                              hipStream_t stream) {
    const float* inputs = (const float*)d_in[0];
    // d_in[1] = label (unused by the reference computation)
    const float* emb    = (const float*)d_in[2];

    float* out    = (float*)d_out;
    float* outQ   = out + 1;                            // quantized_st [N,D]
    float* outEnc = out + 1 + (size_t)N_ * D_ + 1;      // encodings   [N,K]

    // workspace carve-up (~35.8 MB total)
    char* ws = (char*)d_ws;
    __bf16* xb     = (__bf16*)(ws);                         // N*DP*2 = 33,554,432
    __bf16* eb     = (__bf16*)(ws + 33554432);              // K*DP*2 =  2,097,152
    float*  enorm  = (float*) (ws + 35651584);              // K*4
    int*    idxArr = (int*)   (ws + 35655680);              // N*4
    int*    hist   = (int*)   (ws + 35721216);              // K*4
    float*  partial= (float*) (ws + 35725312);              // N*4

    cvt_inputs<<<((size_t)N_ * DP) / 256, 256, 0, stream>>>(inputs, xb);
    cvt_emb<<<K_, 256, 0, stream>>>(emb, eb, enorm, hist);
    vq_argmin<<<N_ / 64, 128, 0, stream>>>(xb, eb, enorm, idxArr);
    vq_epilogue<<<N_, 256, 0, stream>>>(inputs, emb, idxArr, outQ, outEnc, partial, hist);
    vq_final<<<1, 1024, 0, stream>>>(partial, hist, out);
}
